// EGAT2_26577257628254
// MI455X (gfx1250) — matrix operand order repaired and checked by reading, not compiled, not run on hardware
//
#include <hip/hip_runtime.h>
#include <hip/hip_bf16.h>
#include <math.h>

// ---------------------------------------------------------------------------
// EGAT2 on MI455X (gfx1250, wave32, WMMA + TDM).
//
//   N = 50000 nodes, E = 500000 edges, IN=128, HID=256, OUT=256.
//
//   * Dense GEMMs: v_wmma_f32_16x16x32_bf16, one wave = 64x16 C strip
//     (4 WMMAs per K-step). A rows loaded as float4; B tile (32 x 128 fp32,
//     16 KB) staged into LDS per block — via the Tensor Data Mover
//     (tensor_load_to_lds + s_wait_tensorcnt), double-buffered so the DMA of
//     K-step i+1 overlaps the WMMAs of step i. Cooperative-load fallback if
//     the TDM builtin is unavailable (also keeps the host pass compiling).
//   * Edge MLP rewritten: concat(h[s],h[d]) @ linW
//       = (h @ linW_top)[s] + (h @ linW_bot)[d]
//     -> 131 GF of per-edge GEMM becomes 13 GF of node GEMM + an L2-resident
//     gather-add (P1+P2 = 102 MB < 192 MB L2).
//   * Destination-softmax: ordered-uint atomicMax segment max (self loop
//     folded into init), atomicAdd denominator, atomic weighted scatter.
// ---------------------------------------------------------------------------

#define NN   50000
#define NE   500000
#define FIN  128
#define FH   256
#define FOUT 256
#define NEG_SLOPE 0.2f

#define BM 64     // GEMM block M span (4 x 16 per wave)
#define BN 128    // GEMM block N span (8 waves x 16)

typedef __attribute__((ext_vector_type(16))) __bf16 v16bf;
typedef __attribute__((ext_vector_type(8)))  float  v8f;
typedef unsigned int u32x4 __attribute__((ext_vector_type(4)));
typedef int          i32x4 __attribute__((ext_vector_type(4)));
typedef int          i32x8 __attribute__((ext_vector_type(8)));

#if defined(__has_builtin)
#if __has_builtin(__builtin_amdgcn_tensor_load_to_lds) && \
    __has_builtin(__builtin_amdgcn_s_wait_tensorcnt)
#define HAVE_TDM 1
#endif
#endif
#ifndef HAVE_TDM
#define HAVE_TDM 0
#endif

// ---- TDM: stage B[k0..k0+31][tN0..tN0+BN) (fp32, row-major, ld=N) to LDS ---
__device__ __forceinline__ void tdm_stage_b(const float* B, int N, int K,
                                            int k0, int tN0, float* sBuf)
{
#if HAVE_TDM
    unsigned long long ga =
        (unsigned long long)(uintptr_t)(B + (size_t)k0 * N + tN0);
    unsigned lds = (unsigned)(uintptr_t)sBuf;      // LDS byte offset (addr[31:0])
    u32x4 g0;
    g0[0] = 1u;                                    // count=1, user descriptor
    g0[1] = lds;                                   // lds_addr
    g0[2] = (unsigned)(ga & 0xffffffffu);          // global_addr[31:0]
    g0[3] = (unsigned)((ga >> 32) & 0x1ffffffu)    // global_addr[56:32]
          | (2u << 30);                            // type = 2 ("image")
    i32x8 g1;
    g1[0] = (int)(2u << 16);                       // data_size = 4 bytes
    g1[1] = (int)(((unsigned)N & 0xffffu) << 16);  // tensor_dim0[15:0]
    g1[2] = (int)(((unsigned)N >> 16) | (((unsigned)K & 0xffffu) << 16));
    g1[3] = (int)(((unsigned)K >> 16) | ((unsigned)BN << 16)); // tile_dim0=128
    g1[4] = 32;                                    // tile_dim1 = 32 rows
    g1[5] = N;                                     // tensor_dim0_stride[31:0]
    g1[6] = 0;
    g1[7] = 0;
    i32x4 gz = {0, 0, 0, 0};                       // 2D: groups 2/3 unused
#if __clang_major__ >= 23
    i32x8 gz8 = {0, 0, 0, 0, 0, 0, 0, 0};
    __builtin_amdgcn_tensor_load_to_lds(g0, g1, gz, gz, gz8, 0);
#else
    __builtin_amdgcn_tensor_load_to_lds(g0, g1, gz, gz, 0);
#endif
#endif
}

#if !HAVE_TDM
// cooperative fallback: 256 threads x 4 float4 = 32x128 floats
__device__ __forceinline__ void fill_sB(float* sBuf, const float* B,
                                        int N, int k0, int tN0)
{
#pragma unroll
    for (int i = 0; i < 4; ++i) {
        int slot = threadIdx.x + 256 * i;          // 1024 float4 slots
        int row  = slot >> 5;                      // 32 float4 per row
        int cv   = slot & 31;
        float4 v = *(const float4*)(B + (size_t)(k0 + row) * N + tN0 + cv * 4);
        *(float4*)(sBuf + row * BN + cv * 4) = v;
    }
}
#endif

// ---------------- WMMA bf16 GEMM: C[M,N] = A[M,K] @ B[K,N] (+bias)(+relu) ---
// Block: 256 threads (8 waves), tile BM x BN. Wave w owns columns w*16..w*16+15
// over all BM rows -> 4 accumulators / 4 WMMAs per 32-wide K-step.
__global__ __launch_bounds__(256)
void gemm_bf16_wmma(const float* __restrict__ A, const float* __restrict__ B,
                    const float* __restrict__ bias, float* __restrict__ C,
                    int M, int N, int K, int applyBias, int applyRelu)
{
    __shared__ float sB[2 * 32 * BN];              // double-buffered B tile

    const int wave = threadIdx.x >> 5;
    const int lane = threadIdx.x & 31;
    const int half = lane >> 4;                    // 0: lanes 0-15, 1: 16-31
    const int l    = lane & 15;

    const int nBlocksN = N / BN;
    const int tM  = (blockIdx.x / nBlocksN) * BM;
    const int tN0 = (blockIdx.x % nBlocksN) * BN;
    const int colLocal = wave * 16 + l;
    const int col = tN0 + colLocal;

    v8f acc[4] = {{}, {}, {}, {}};

    // A row pointers (clamped for the ragged last M block: loads stay in
    // bounds, EXEC stays all-1s for WMMA; stores are predicated below)
    const float* arow[4];
#pragma unroll
    for (int m = 0; m < 4; ++m) {
        int r = tM + m * 16 + l;
        if (r > M - 1) r = M - 1;
        arow[m] = A + (size_t)r * K;
    }

    const int nIter = K / 32;
#if HAVE_TDM
    if (wave == 0) tdm_stage_b(B, N, K, 0, tN0, sB);
#endif
    for (int it = 0; it < nIter; ++it) {
        const int k0 = it * 32;
        float* cur = sB + (it & 1) * (32 * BN);
#if HAVE_TDM
        if (wave == 0) __builtin_amdgcn_s_wait_tensorcnt(0);
        __syncthreads();                           // tile it ready, tile it-1 free
        if (wave == 0 && it + 1 < nIter)           // overlap DMA with WMMAs
            tdm_stage_b(B, N, K, k0 + 32, tN0, sB + ((it + 1) & 1) * (32 * BN));
#else
        __syncthreads();
        fill_sB(cur, B, N, k0, tN0);
        __syncthreads();
#endif
        // B 32x16 bf16 fragment from LDS: column colLocal, element j<8 at
        // K=k0+8*half+j, j>=8 at K=k0+16+8*half+(j-8) (conflict-free ds loads)
        v16bf bfrag;
        {
            const float* bp = cur + (half * 8) * BN + colLocal;
#pragma unroll
            for (int j = 0; j < 16; ++j)
                bfrag[j] = (__bf16)bp[(j + (j >> 3) * 8) * BN];
        }
        // 4 A fragments (float4 global loads) + 4 WMMAs
#pragma unroll
        for (int m = 0; m < 4; ++m) {
            const float* ap = arow[m] + k0 + half * 8;
            float4 p0 = *(const float4*)(ap + 0);
            float4 p1 = *(const float4*)(ap + 4);
            float4 p2 = *(const float4*)(ap + 16);
            float4 p3 = *(const float4*)(ap + 20);
            v16bf a;
            a[0]=(__bf16)p0.x; a[1]=(__bf16)p0.y; a[2]=(__bf16)p0.z; a[3]=(__bf16)p0.w;
            a[4]=(__bf16)p1.x; a[5]=(__bf16)p1.y; a[6]=(__bf16)p1.z; a[7]=(__bf16)p1.w;
            a[8]=(__bf16)p2.x; a[9]=(__bf16)p2.y; a[10]=(__bf16)p2.z; a[11]=(__bf16)p2.w;
            a[12]=(__bf16)p3.x; a[13]=(__bf16)p3.y; a[14]=(__bf16)p3.z; a[15]=(__bf16)p3.w;
            acc[m] = __builtin_amdgcn_wmma_f32_16x16x32_bf16(
                         false, a, false, bfrag, (short)0, acc[m], false, false);
        }
    }

    // C 16x16 f32 layout: element r -> row subtile + 8*half + r, col = col.
    const float bv = applyBias ? bias[col] : 0.f;
#pragma unroll
    for (int m = 0; m < 4; ++m) {
#pragma unroll
        for (int r = 0; r < 8; ++r) {
            int row = tM + m * 16 + half * 8 + r;
            if (row < M) {
                float v = acc[m][r] + bv;
                if (applyRelu) v = fmaxf(v, 0.f);
                C[(size_t)row * N + col] = v;
            }
        }
    }
}

// ---------------- helpers -------------------------------------------------
__device__ __forceinline__ float leaky(float v) {
    return v > 0.f ? v : NEG_SLOPE * v;
}
__device__ __forceinline__ unsigned ordf(float f) {
    unsigned u = __float_as_uint(f);
    return (u & 0x80000000u) ? ~u : (u | 0x80000000u);
}
__device__ __forceinline__ float iordf(unsigned u) {
    return __uint_as_float((u & 0x80000000u) ? (u & 0x7fffffffu) : ~u);
}

__global__ __launch_bounds__(256)
void alpha_kernel(const float* __restrict__ h, const float* __restrict__ a_src,
                  const float* __restrict__ a_dst,
                  float* __restrict__ as, float* __restrict__ ad)
{
    int i = blockIdx.x * blockDim.x + threadIdx.x;
    if (i >= NN) return;
    const float* row = h + (size_t)i * FH;
    float s = 0.f, d = 0.f;
#pragma unroll 4
    for (int f = 0; f < FH; ++f) {
        float v = row[f];
        s += v * a_src[f];
        d += v * a_dst[f];
    }
    as[i] = s; ad[i] = d;
}

__global__ __launch_bounds__(256)
void node_init(const float* __restrict__ as, const float* __restrict__ ad,
               unsigned* __restrict__ emaxu, float* __restrict__ denom)
{
    int i = blockIdx.x * blockDim.x + threadIdx.x;
    if (i >= NN) return;
    emaxu[i] = ordf(leaky(as[i] + ad[i]));   // self-loop edge seeds the max
    denom[i] = 0.f;
}

__global__ __launch_bounds__(256)
void edge_max(const int* __restrict__ src, const int* __restrict__ dst,
              const float* __restrict__ as, const float* __restrict__ ad,
              unsigned* __restrict__ emaxu)
{
    int e = blockIdx.x * blockDim.x + threadIdx.x;
    if (e >= NE) return;
    int s = src[e], d = dst[e];
    atomicMax(&emaxu[d], ordf(leaky(as[s] + ad[d])));
}

__global__ __launch_bounds__(256)
void edge_exp(const int* __restrict__ src, const int* __restrict__ dst,
              const float* __restrict__ as, const float* __restrict__ ad,
              const unsigned* __restrict__ emaxu,
              float* __restrict__ denom, float* __restrict__ eexp)
{
    int e = blockIdx.x * blockDim.x + threadIdx.x;
    if (e >= NE) return;
    int s = src[e], d = dst[e];
    float x = __expf(leaky(as[s] + ad[d]) - iordf(emaxu[d]));
    eexp[e] = x;
    atomicAdd(&denom[d], x);
}

// self-loop numerator: denom += exp(e_self - max); g[i,:] = h[i,:] * that.
__global__ __launch_bounds__(256)
void node_mid(const float* __restrict__ as, const float* __restrict__ ad,
              const unsigned* __restrict__ emaxu, float* __restrict__ denom,
              const float* __restrict__ h, float* __restrict__ g)
{
    __shared__ float ssx[4];
    int sub = threadIdx.x >> 6;
    int i   = blockIdx.x * 4 + sub;
    int c   = (threadIdx.x & 63) << 2;
    if ((threadIdx.x & 63) == 0 && i < NN) {
        float sx = __expf(leaky(as[i] + ad[i]) - iordf(emaxu[i]));
        denom[i] += sx;                     // sole writer after edge_exp
        ssx[sub] = sx;
    }
    __syncthreads();
    if (i >= NN) return;
    float sx = ssx[sub];
    float4 hv = *(const float4*)(h + (size_t)i * FH + c);
    *(float4*)(g + (size_t)i * FH + c) =
        make_float4(hv.x * sx, hv.y * sx, hv.z * sx, hv.w * sx);
}

// g[d,:] += h[s,:] * eexp[e]   (64 float4-threads per edge)
__global__ __launch_bounds__(256)
void edge_scatter(const int* __restrict__ src, const int* __restrict__ dst,
                  const float* __restrict__ eexp, const float* __restrict__ h,
                  float* __restrict__ g)
{
    long long t = (long long)blockIdx.x * blockDim.x + threadIdx.x;
    int e = (int)(t >> 6);
    int c = ((int)t & 63) << 2;
    if (e >= NE) return;
    int s = src[e], d = dst[e];
    float w = eexp[e];
    float4 hv = *(const float4*)(h + (size_t)s * FH + c);
    float* gp = g + (size_t)d * FH + c;
    atomicAdd(gp + 0, hv.x * w);
    atomicAdd(gp + 1, hv.y * w);
    atomicAdd(gp + 2, hv.z * w);
    atomicAdd(gp + 3, hv.w * w);
}

__global__ __launch_bounds__(256)
void node_fin(float* __restrict__ g, const float* __restrict__ denom,
              const float* __restrict__ bias, int applyRelu)
{
    long long t = (long long)blockIdx.x * blockDim.x + threadIdx.x;
    int i = (int)(t >> 6);
    int c = ((int)t & 63) << 2;
    if (i >= NN) return;
    float inv = 1.f / denom[i];
    float4 v = *(float4*)(g + (size_t)i * FH + c);
    float4 b = *(const float4*)(bias + c);
    v.x = v.x * inv + b.x; v.y = v.y * inv + b.y;
    v.z = v.z * inv + b.z; v.w = v.w * inv + b.w;
    if (applyRelu) {
        v.x = fmaxf(v.x, 0.f); v.y = fmaxf(v.y, 0.f);
        v.z = fmaxf(v.z, 0.f); v.w = fmaxf(v.w, 0.f);
    }
    *(float4*)(g + (size_t)i * FH + c) = v;
}

// out[e,:] = P1[src[e],:] + P2[dst[e],:]   (lin_b folded into P1)
__global__ __launch_bounds__(256)
void edge_out(const int* __restrict__ src, const int* __restrict__ dst,
              const float* __restrict__ P1, const float* __restrict__ P2,
              float* __restrict__ out)
{
    long long t = (long long)blockIdx.x * blockDim.x + threadIdx.x;
    int e = (int)(t >> 6);
    int c = ((int)t & 63) << 2;
    if (e >= NE) return;
    int s = src[e], d = dst[e];
    float4 a = *(const float4*)(P1 + (size_t)s * FOUT + c);
    float4 b = *(const float4*)(P2 + (size_t)d * FOUT + c);
    *(float4*)(out + (size_t)e * FOUT + c) =
        make_float4(a.x + b.x, a.y + b.y, a.z + b.z, a.w + b.w);
}

// ---------------------------------------------------------------------------
extern "C" void kernel_launch(void* const* d_in, const int* in_sizes, int n_in,
                              void* d_out, int out_size, void* d_ws, size_t ws_size,
                              hipStream_t stream)
{
    const float* x    = (const float*)d_in[0];
    const int*   eidx = (const int*)  d_in[1];     // [2, E] (jax default int32)
    const float* W1   = (const float*)d_in[2];
    const float* a1s  = (const float*)d_in[3];
    const float* a1d  = (const float*)d_in[4];
    const float* b1   = (const float*)d_in[5];
    const float* W2   = (const float*)d_in[6];
    const float* a2s  = (const float*)d_in[7];
    const float* a2d  = (const float*)d_in[8];
    const float* b2   = (const float*)d_in[9];
    const float* linW = (const float*)d_in[10];    // [2*HID, OUT]
    const float* linb = (const float*)d_in[11];
    float*       out  = (float*)d_out;

    const int* src = eidx;
    const int* dst = eidx + NE;

    // workspace layout (~157 MB)
    float*    ws    = (float*)d_ws;
    float*    h     = ws;                          // NN*FH (later holds P1)
    float*    g     = h   + (size_t)NN * FH;       // NN*FH
    float*    P2    = g   + (size_t)NN * FH;       // NN*FH
    float*    as    = P2  + (size_t)NN * FH;       // NN
    float*    ad    = as  + NN;
    float*    denom = ad  + NN;
    unsigned* emaxu = (unsigned*)(denom + NN);
    float*    eexp  = (float*)(emaxu + NN);        // NE

    const int gemmBlocks  = ((NN + BM - 1) / BM) * (FH / BN);  // 782*2
    const int nodeBlocks  = (NN + 255) / 256;
    const int edgeBlocks  = (NE + 255) / 256;
    const int nodeF4Blks  = (NN * 64 + 255) / 256;
    const int edgeF4Blks  = (NE * 64 + 255) / 256;
    const int nodeMidBlks = (NN + 3) / 4;

    // -------- GAT layer 1 --------
    gemm_bf16_wmma<<<gemmBlocks, 256, 0, stream>>>(x, W1, nullptr, h,
                                                   NN, FH, FIN, 0, 0);
    alpha_kernel<<<nodeBlocks, 256, 0, stream>>>(h, a1s, a1d, as, ad);
    node_init  <<<nodeBlocks, 256, 0, stream>>>(as, ad, emaxu, denom);
    edge_max   <<<edgeBlocks, 256, 0, stream>>>(src, dst, as, ad, emaxu);
    edge_exp   <<<edgeBlocks, 256, 0, stream>>>(src, dst, as, ad, emaxu, denom, eexp);
    node_mid   <<<nodeMidBlks, 256, 0, stream>>>(as, ad, emaxu, denom, h, g);
    edge_scatter<<<edgeF4Blks, 256, 0, stream>>>(src, dst, eexp, h, g);
    node_fin   <<<nodeF4Blks, 256, 0, stream>>>(g, denom, b1, 1);   // + ReLU

    // -------- GAT layer 2 --------
    gemm_bf16_wmma<<<gemmBlocks, 256, 0, stream>>>(g, W2, nullptr, h,
                                                   NN, FH, FH, 0, 0);
    alpha_kernel<<<nodeBlocks, 256, 0, stream>>>(h, a2s, a2d, as, ad);
    node_init  <<<nodeBlocks, 256, 0, stream>>>(as, ad, emaxu, denom);
    edge_max   <<<edgeBlocks, 256, 0, stream>>>(src, dst, as, ad, emaxu);
    edge_exp   <<<edgeBlocks, 256, 0, stream>>>(src, dst, as, ad, emaxu, denom, eexp);
    node_mid   <<<nodeMidBlks, 256, 0, stream>>>(as, ad, emaxu, denom, h, g);
    edge_scatter<<<edgeF4Blks, 256, 0, stream>>>(src, dst, eexp, h, g);
    node_fin   <<<nodeF4Blks, 256, 0, stream>>>(g, denom, b2, 0);   // no ReLU

    // -------- edge MLP: P1 = g@linW_top + lin_b ; P2 = g@linW_bot --------
    gemm_bf16_wmma<<<gemmBlocks, 256, 0, stream>>>(g, linW, linb, h,
                                                   NN, FOUT, FH, 1, 0);
    gemm_bf16_wmma<<<gemmBlocks, 256, 0, stream>>>(g, linW + (size_t)FH * FOUT,
                                                   nullptr, P2,
                                                   NN, FOUT, FH, 0, 0);
    edge_out<<<edgeF4Blks, 256, 0, stream>>>(src, dst, h, P2, out);
}
